// Transformer_34737695490197
// MI455X (gfx1250) — matrix operand order repaired and checked by reading, not compiled, not run on hardware
//
#include <hip/hip_runtime.h>
#include <cstdint>
#include <cstddef>

// ---------------- problem constants (match reference) ----------------
#define BB 4
#define SS 1024
#define HH 16
#define DD 1024
#define LL 6
#define VV 256
#define DHD 64
#define DFF 4096

// ---------------- CDNA5 WMMA types ----------------
typedef __attribute__((ext_vector_type(16))) __bf16        v16bf;
typedef __attribute__((ext_vector_type(8)))  float         v8f;
typedef __attribute__((ext_vector_type(4)))  unsigned int  v4u;

// trivially-constructible fragment: fill via two 128-bit LDS loads
union BF16x16 {
  v16bf v;
  v4u   q[2];
  unsigned short u[16];
};

__device__ __forceinline__ unsigned short f2bf(float f) {
  union { float f; uint32_t u; } c; c.f = f;
  uint32_t x = c.u;
  if ((x & 0x7f800000u) == 0x7f800000u) return (unsigned short)(x >> 16); // inf/nan passthrough
  return (unsigned short)((x + 0x7fffu + ((x >> 16) & 1u)) >> 16);        // RNE
}
__device__ __forceinline__ float bf2f(unsigned short b) {
  union { uint32_t u; float f; } c; c.u = ((uint32_t)b) << 16;
  return c.f;
}
__device__ __forceinline__ v8f wmma_bf16(v16bf a, v16bf b, v8f c) {
  // D = A(16x32 bf16) * B(32x16 bf16) + C(16x16 f32)
  return __builtin_amdgcn_wmma_f32_16x16x32_bf16(false, a, false, b, (short)0, c, false, false);
}

// ---- CDNA5 async global->LDS copy (GLOBAL_LOAD_ASYNC_TO_LDS_B128, ASYNCcnt) ----
__device__ __forceinline__ void async_g2l_b128(void* ldsp, const void* g) {
  unsigned lds_off = (unsigned)(uintptr_t)ldsp;               // addr[31:0] = LDS offset
  unsigned long long ga = (unsigned long long)(uintptr_t)g;
  asm volatile("global_load_async_to_lds_b128 %0, %1, off"
               :: "v"(lds_off), "v"(ga) : "memory");
}
__device__ __forceinline__ void async_wait0() {
  asm volatile("s_wait_asynccnt 0x0" ::: "memory");
}

// ---------------- tiled fp32->bf16 transpose-convert: dst[n][k] = bf16(src[k][n]) ----------------
// grid (N/32, K/32, layers), block 256
__global__ __launch_bounds__(256) void transpose_convert(
    const float* __restrict__ src, unsigned short* __restrict__ dst, int K, int N) {
  __shared__ float t[32][33];
  const size_t mo = (size_t)blockIdx.z * K * N;
  const int k0 = blockIdx.y * 32, n0 = blockIdx.x * 32;
  const int tx = threadIdx.x & 31, ty = threadIdx.x >> 5;
  for (int j = 0; j < 32; j += 8)
    t[ty + j][tx] = src[mo + (size_t)(k0 + ty + j) * N + n0 + tx];
  __syncthreads();
  for (int j = 0; j < 32; j += 8)
    dst[mo + (size_t)(n0 + ty + j) * K + k0 + tx] = f2bf(t[tx][ty + j]);
}

// ---------------- embedding + positional encoding ----------------
__global__ void embed_kernel(const int* __restrict__ tokens,
                             const float* __restrict__ emb,
                             float* __restrict__ x,
                             unsigned short* __restrict__ xb) {
  size_t idx = (size_t)blockIdx.x * blockDim.x + threadIdx.x;
  if (idx >= (size_t)BB * SS * DD) return;
  int d  = (int)(idx % DD);
  size_t row = idx / DD;              // b*S + s
  int s  = (int)(row % SS);
  int tok = tokens[row];
  float pos = (float)s / 10000.0f;
  float power = 2.0f * (float)d / (float)DD;
  float pe = powf(pos, power);        // powf(0,0)==1 matches jnp
  pe = (d & 1) ? cosf(pe) : sinf(pe);
  float v = emb[(size_t)tok * DD + d] + pe;
  x[idx]  = v;
  xb[idx] = f2bf(v);
}

// ---------------- WMMA GEMM: C[M,N] = A[M,K](bf16) @ Bt[N,K](bf16, pre-transposed) + bias ----------------
// 256 threads (8 waves), tile 128x128, K staged by 64, double-buffered async global->LDS staging.
// Wave w: rows (w&3)*32, cols (w>>2)*64 (2x4 f32 accumulators).
// Requires: M%128==0, N%128==0, K%64==0 (true for all GEMMs here).
#define LAP 72   // lA row stride (64 + 8 pad), 144B (16B aligned)
#define LBP 72   // lBt row stride
#define GEMM_BUF (128 * LAP + 128 * LBP)   // ushorts per buffer

__global__ __launch_bounds__(256) void gemm_bf16(
    const unsigned short* __restrict__ A,
    const unsigned short* __restrict__ Bt,   // [N][K]
    const float* __restrict__ bias,
    float* __restrict__ outF,
    unsigned short* __restrict__ outB,
    int M, int N, int K, int relu, int transOutB) {
  extern __shared__ unsigned short gsm[];   // 2 * GEMM_BUF ushorts

  const int tid  = threadIdx.x;
  const int lane = tid & 31;
  const int wave = tid >> 5;
  const int mw = (wave & 3) * 32;
  const int nw = (wave >> 2) * 64;
  const int tileM = blockIdx.y * 128;
  const int tileN = blockIdx.x * 128;
  const int hl = (lane < 16) ? 0 : 1;        // half-wave select
  const int ln = lane & 15;
  const int kbA = hl * 16;                   // A frag K base
  const int kbB = hl * 16;                   // B frag K base

  v8f zero = {};
  v8f acc[2][4];
  for (int i = 0; i < 2; ++i)
    for (int j = 0; j < 4; ++j) acc[i][j] = zero;

  const int sr = tid >> 1, sc = (tid & 1) * 32;   // staging: 32 elems/thread (both tiles)

  auto stage = [&](int k0, unsigned short* buf) {
    unsigned short* lA  = buf;
    unsigned short* lBt = buf + 128 * LAP;
    const unsigned short* ga = &A [(size_t)(tileM + sr) * K + k0 + sc];
    const unsigned short* gb = &Bt[(size_t)(tileN + sr) * K + k0 + sc];
    #pragma unroll
    for (int j = 0; j < 4; ++j) {
      async_g2l_b128(&lA [sr * LAP + sc + j * 8], ga + j * 8);
      async_g2l_b128(&lBt[sr * LBP + sc + j * 8], gb + j * 8);
    }
  };

  stage(0, gsm);
  int ib = 0;
  for (int k0 = 0; k0 < K; k0 += 64, ib ^= 1) {
    async_wait0();            // this wave's copies for stage k0 have landed
    __syncthreads();          // everyone's copies have landed; prev buffer free
    if (k0 + 64 < K) stage(k0 + 64, gsm + (ib ^ 1) * GEMM_BUF);

    const unsigned short* lA  = gsm + ib * GEMM_BUF;
    const unsigned short* lBt = lA + 128 * LAP;
    #pragma unroll
    for (int kk = 0; kk < 64; kk += 32) {
      BF16x16 af[2];
      #pragma unroll
      for (int rt = 0; rt < 2; ++rt) {
        const unsigned short* p = &lA[(mw + rt * 16 + ln) * LAP + kk];
        af[rt].q[0] = *reinterpret_cast<const v4u*>(p + kbA);
        af[rt].q[1] = *reinterpret_cast<const v4u*>(p + 8 + kbA);
      }
      #pragma unroll
      for (int ct = 0; ct < 4; ++ct) {
        BF16x16 bfm;
        const unsigned short* p = &lBt[(nw + ct * 16 + ln) * LBP + kk + kbB];
        bfm.q[0] = *reinterpret_cast<const v4u*>(p);
        bfm.q[1] = *reinterpret_cast<const v4u*>(p + 8);
        #pragma unroll
        for (int rt = 0; rt < 2; ++rt)
          acc[rt][ct] = wmma_bf16(af[rt].v, bfm.v, acc[rt][ct]);
      }
    }
  }

  // epilogue: bias (+ReLU); store fp32 and/or bf16 (optionally transposed bf16 [n][m])
  for (int rt = 0; rt < 2; ++rt) {
    for (int ct = 0; ct < 4; ++ct) {
      const int n  = tileN + nw + ct * 16 + ln;
      const int m0 = tileM + mw + rt * 16 + hl * 8;
      const float bv = bias ? bias[n] : 0.0f;
      for (int r = 0; r < 8; ++r) {
        float v = acc[rt][ct][r] + bv;
        if (relu) v = fmaxf(v, 0.0f);
        if (outF) outF[(size_t)(m0 + r) * N + n] = v;
        if (outB) {
          if (transOutB) outB[(size_t)n * M + m0 + r] = f2bf(v);  // [n][m]
          else           outB[(size_t)(m0 + r) * N + n] = f2bf(v);
        }
      }
    }
  }
}

// ---------------- fused multi-head attention ----------------
// grid = (S/64, H, B); block = 128 (4 waves); wave w owns 16 query rows.
// Full 64x1024 bf16 score rows in LDS (CDNA5: 320KB/WGP); K/V chunks double-buffered async.
#define SP 1032   // score row stride (ushorts), 2064B
#define QP 72     // Q / K-chunk row stride
#define VP 40     // V-chunk row stride ([d][key], 32 + 8 pad)
#define KVBUF 2560  // max(32*QP, 64*VP) ushorts per buffer

__global__ __launch_bounds__(128) void attention_kernel(
    const unsigned short* __restrict__ Q,    // [B*S][D]
    const unsigned short* __restrict__ Kd,   // [B*S][D]
    const unsigned short* __restrict__ Vt,   // [D][B*S]  (pre-transposed by V GEMM)
    float* __restrict__ Y, float scale) {
  extern __shared__ unsigned short smem[];
  unsigned short* scoresB = smem;                    // [64][SP]
  unsigned short* ldsQ    = smem + 64 * SP;          // [64][QP]
  unsigned short* ldsKV   = ldsQ + 64 * QP;          // 2 * KVBUF

  __shared__ float redm[128];  // softmax split-row combine

  const int tid  = threadIdx.x;
  const int lane = tid & 31;
  const int wave = tid >> 5;
  const int hl = (lane < 16) ? 0 : 1;
  const int ln = lane & 15;
  const int kbA = hl * 16, kbB = hl * 16;
  const int q0 = blockIdx.x * 64;
  const int h  = blockIdx.y;
  const int b  = blockIdx.z;
  const size_t baseQ = ((size_t)b * SS + q0) * DD + h * DHD;
  const size_t baseK = ((size_t)b * SS) * DD + h * DHD;

  // async-load Q tile 64x64
  {
    const int r = tid >> 1, c0 = (tid & 1) * 32;
    const unsigned short* g = &Q[baseQ + (size_t)r * DD + c0];
    #pragma unroll
    for (int j = 0; j < 4; ++j)
      async_g2l_b128(&ldsQ[r * QP + c0 + j * 8], g + j * 8);
  }

  const int kvr = tid >> 2, kvc = (tid & 3) * 16;    // K-chunk staging: 16 elems/thread
  auto stageK = [&](int kt, unsigned short* buf) {
    const unsigned short* g = &Kd[baseK + (size_t)(kt + kvr) * DD + kvc];
    async_g2l_b128(&buf[kvr * QP + kvc],     g);
    async_g2l_b128(&buf[kvr * QP + kvc + 8], g + 8);
  };
  const int vr = tid >> 1, vc = (tid & 1) * 16;      // V-chunk staging: [d][key]
  auto stageV = [&](int kt, unsigned short* buf) {
    const unsigned short* g = &Vt[(size_t)(h * DHD + vr) * (BB * SS) + (size_t)b * SS + kt + vc];
    async_g2l_b128(&buf[vr * VP + vc],     g);
    async_g2l_b128(&buf[vr * VP + vc + 8], g + 8);
  };

  // -------- phase A: scores = scale * Q @ K^T (double-buffered K chunks) --------
  stageK(0, ldsKV);
  int ib = 0;
  for (int kt = 0; kt < SS; kt += 32, ib ^= 1) {
    async_wait0();
    __syncthreads();
    if (kt + 32 < SS) stageK(kt + 32, ldsKV + (ib ^ 1) * KVBUF);
    const unsigned short* kbuf = ldsKV + ib * KVBUF;

    v8f zero = {};
    v8f scv[2]; scv[0] = zero; scv[1] = zero;
    const int qrow = wave * 16 + ln;
    #pragma unroll
    for (int kk = 0; kk < DHD; kk += 32) {
      BF16x16 af;
      {
        const unsigned short* p = &ldsQ[qrow * QP + kk];
        af.q[0] = *reinterpret_cast<const v4u*>(p + kbA);
        af.q[1] = *reinterpret_cast<const v4u*>(p + 8 + kbA);
      }
      #pragma unroll
      for (int ct = 0; ct < 2; ++ct) {
        BF16x16 bfm;  // B[k=d][n=key] = kbuf[n][d]: contiguous along d
        const unsigned short* p = &kbuf[(ct * 16 + ln) * QP + kk + kbB];
        bfm.q[0] = *reinterpret_cast<const v4u*>(p);
        bfm.q[1] = *reinterpret_cast<const v4u*>(p + 8);
        scv[ct] = wmma_bf16(af.v, bfm.v, scv[ct]);
      }
    }
    #pragma unroll
    for (int ct = 0; ct < 2; ++ct) {
      const int col = kt + ct * 16 + ln;
      const int rbase = wave * 16 + hl * 8;
      #pragma unroll
      for (int r = 0; r < 8; ++r)
        scoresB[(rbase + r) * SP + col] = f2bf(scv[ct][r] * scale);
    }
  }
  __syncthreads();

  // -------- phase B: row softmax, 2 threads per row --------
  {
    const int row = tid & 63, half = tid >> 6;       // half: cols [half*512, +512)
    unsigned short* rp = &scoresB[row * SP + half * 512];
    float mx = -3.4e38f;
    for (int c = 0; c < 512; ++c) mx = fmaxf(mx, bf2f(rp[c]));
    redm[row * 2 + half] = mx;
    __syncthreads();
    mx = fmaxf(redm[row * 2], redm[row * 2 + 1]);
    float sum = 0.0f;
    for (int c = 0; c < 512; ++c) {
      float e = __expf(bf2f(rp[c]) - mx);
      sum += e;
      rp[c] = f2bf(e);
    }
    __syncthreads();                                  // redm reuse
    redm[row * 2 + half] = sum;
    __syncthreads();
    const float inv = 1.0f / (redm[row * 2] + redm[row * 2 + 1]);
    for (int c = 0; c < 512; ++c) rp[c] = f2bf(bf2f(rp[c]) * inv);
  }
  __syncthreads();

  // -------- phase C: Y = P @ V (double-buffered V chunks from Vt) --------
  v8f zero = {};
  v8f yacc[4];
  for (int j = 0; j < 4; ++j) yacc[j] = zero;
  const int qrow = wave * 16 + ln;
  stageV(0, ldsKV);
  ib = 0;
  for (int kt = 0; kt < SS; kt += 32, ib ^= 1) {
    async_wait0();
    __syncthreads();
    if (kt + 32 < SS) stageV(kt + 32, ldsKV + (ib ^ 1) * KVBUF);
    const unsigned short* vbuf = ldsKV + ib * KVBUF;

    BF16x16 af;  // A = P rows, K = key index (contiguous in scoresB)
    {
      const unsigned short* p = &scoresB[qrow * SP + kt];
      af.q[0] = *reinterpret_cast<const v4u*>(p + kbA);
      af.q[1] = *reinterpret_cast<const v4u*>(p + 8 + kbA);
    }
    #pragma unroll
    for (int ct = 0; ct < 4; ++ct) {
      BF16x16 bfm;  // B[k=key][n=d] = vbuf[n][key]: contiguous along key
      const unsigned short* p = &vbuf[(ct * 16 + ln) * VP + kbB];
      bfm.q[0] = *reinterpret_cast<const v4u*>(p);
      bfm.q[1] = *reinterpret_cast<const v4u*>(p + 8);
      yacc[ct] = wmma_bf16(af.v, bfm.v, yacc[ct]);
    }
  }
  for (int ct = 0; ct < 4; ++ct) {
    const int col = h * DHD + ct * 16 + ln;
    const int rbase = q0 + wave * 16 + hl * 8;
    for (int r = 0; r < 8; ++r)
      Y[((size_t)b * SS + rbase + r) * DD + col] = yacc[ct][r];
  }
}

// ---------------- residual add + LayerNorm (wave32 reductions) ----------------
__global__ __launch_bounds__(256) void add_layernorm(
    const float* __restrict__ xin, const float* __restrict__ y,
    float* __restrict__ xout, unsigned short* __restrict__ xb) {
  __shared__ float rs[8], rs2[8];
  const size_t row = blockIdx.x;
  const float* xr = xin + row * DD;
  const float* yr = y + row * DD;
  float v[4], s = 0.0f, s2 = 0.0f;
  for (int i = 0; i < 4; ++i) {
    const int c = threadIdx.x + i * 256;
    v[i] = xr[c] + yr[c];
    s += v[i]; s2 += v[i] * v[i];
  }
  for (int off = 16; off > 0; off >>= 1) {
    s  += __shfl_xor(s, off, 32);
    s2 += __shfl_xor(s2, off, 32);
  }
  if ((threadIdx.x & 31) == 0) { rs[threadIdx.x >> 5] = s; rs2[threadIdx.x >> 5] = s2; }
  __syncthreads();
  float ts = 0.0f, ts2 = 0.0f;
  for (int i = 0; i < 8; ++i) { ts += rs[i]; ts2 += rs2[i]; }
  const float mean = ts / (float)DD;
  const float var  = ts2 / (float)DD - mean * mean;
  const float rstd = rsqrtf(var + 1e-5f);
  for (int i = 0; i < 4; ++i) {
    const int c = threadIdx.x + i * 256;
    const float o = (v[i] - mean) * rstd;
    xout[row * DD + c] = o;
    xb[row * DD + c]   = f2bf(o);
  }
}

// ---------------- final vocab softmax (V=256, one thread per column) ----------------
__global__ __launch_bounds__(256) void softmax_out(const float* __restrict__ logits,
                                                   float* __restrict__ out) {
  __shared__ float sm[8], ss[8];
  const size_t row = blockIdx.x;
  const int t = threadIdx.x;
  float v = logits[row * VV + t];
  float m = v;
  for (int off = 16; off > 0; off >>= 1) m = fmaxf(m, __shfl_xor(m, off, 32));
  if ((t & 31) == 0) sm[t >> 5] = m;
  __syncthreads();
  float bm = sm[0];
  for (int i = 1; i < 8; ++i) bm = fmaxf(bm, sm[i]);
  float e = __expf(v - bm);
  float sum = e;
  for (int off = 16; off > 0; off >>= 1) sum += __shfl_xor(sum, off, 32);
  if ((t & 31) == 0) ss[t >> 5] = sum;
  __syncthreads();
  float bs = 0.0f;
  for (int i = 0; i < 8; ++i) bs += ss[i];
  out[row * VV + t] = e / bs;
}

// ---------------- host orchestration ----------------
extern "C" void kernel_launch(void* const* d_in, const int* in_sizes, int n_in,
                              void* d_out, int out_size, void* d_ws, size_t ws_size,
                              hipStream_t stream) {
  const int*   tokens = (const int*)  d_in[0];
  const float* emb    = (const float*)d_in[1];
  const float* Wq  = (const float*)d_in[2];  const float* bq  = (const float*)d_in[3];
  const float* Wk  = (const float*)d_in[4];  const float* bk  = (const float*)d_in[5];
  const float* Wv  = (const float*)d_in[6];  const float* bv  = (const float*)d_in[7];
  const float* W1  = (const float*)d_in[8];  const float* b1  = (const float*)d_in[9];
  const float* W2  = (const float*)d_in[10]; const float* b2  = (const float*)d_in[11];
  const float* Wout= (const float*)d_in[12]; const float* bout= (const float*)d_in[13];
  float* out = (float*)d_out;

  char* ws = (char*)d_ws;
  size_t off = 0;
  auto carve = [&](size_t bytes) -> char* {
    char* p = ws + off;
    off += (bytes + 255) & ~(size_t)255;
    return p;
  };
  const size_t nDD  = (size_t)LL * DD * DD;
  const size_t nDDF = (size_t)LL * DD * DFF;
  unsigned short* wqT   = (unsigned short*)carve(nDD * 2);     // [l][n][k]
  unsigned short* wkT   = (unsigned short*)carve(nDD * 2);
  unsigned short* wvT   = (unsigned short*)carve(nDD * 2);
  unsigned short* w1T   = (unsigned short*)carve(nDDF * 2);
  unsigned short* w2T   = (unsigned short*)carve(nDDF * 2);
  unsigned short* woutT = (unsigned short*)carve((size_t)DD * VV * 2);
  float*          xF    = (float*)carve((size_t)BB * SS * DD * 4);
  unsigned short* xB    = (unsigned short*)carve((size_t)BB * SS * DD * 2);
  unsigned short* qB    = (unsigned short*)carve((size_t)BB * SS * DD * 2);
  unsigned short* kB    = (unsigned short*)carve((size_t)BB * SS * DD * 2);
  unsigned short* vT    = (unsigned short*)carve((size_t)BB * SS * DD * 2);  // [D][B*S]
  float*          yF    = (float*)carve((size_t)BB * SS * DD * 4);
  unsigned short* h1B   = (unsigned short*)carve((size_t)BB * SS * DFF * 2);
  float*          logit = (float*)carve((size_t)BB * SS * VV * 4);
  (void)ws_size; (void)in_sizes; (void)n_in; (void)out_size;

  // weight transpose-convert (re-done every call: deterministic)
  auto tconv = [&](const float* s, unsigned short* d, int K, int N, int layers) {
    transpose_convert<<<dim3(N / 32, K / 32, layers), dim3(256), 0, stream>>>(s, d, K, N);
  };
  tconv(Wq, wqT, DD, DD, LL); tconv(Wk, wkT, DD, DD, LL); tconv(Wv, wvT, DD, DD, LL);
  tconv(W1, w1T, DD, DFF, LL); tconv(W2, w2T, DFF, DD, LL); tconv(Wout, woutT, DD, VV, 1);

  // x = emb[tokens] + positional encoding
  {
    const size_t n = (size_t)BB * SS * DD;
    embed_kernel<<<dim3((unsigned)((n + 255) / 256)), dim3(256), 0, stream>>>(tokens, emb, xF, xB);
  }

  const int M = BB * SS;  // 4096
  const unsigned gemmSmem = 2 * GEMM_BUF * 2;                   // ~72 KB dynamic LDS
  auto gemm = [&](const unsigned short* A, const unsigned short* Bm, const float* bias,
                  float* oF, unsigned short* oB, int N, int K, int relu, int transB) {
    gemm_bf16<<<dim3(N / 128, M / 128), dim3(256), gemmSmem, stream>>>(
        A, Bm, bias, oF, oB, M, N, K, relu, transB);
  };

  const unsigned attnSmem = (64 * SP + 64 * QP + 2 * KVBUF) * 2;  // ~148 KB dynamic LDS

  for (int l = 0; l < LL; ++l) {
    const size_t wo  = (size_t)l * DD * DD;
    const size_t w1o = (size_t)l * DD * DFF;
    const size_t w2o = (size_t)l * DFF * DD;
    // QKV projections (bf16 out; V stored transposed [d][b*S+s])
    gemm(xB, wqT + wo, bq + (size_t)l * DD, nullptr, qB, DD, DD, 0, 0);
    gemm(xB, wkT + wo, bk + (size_t)l * DD, nullptr, kB, DD, DD, 0, 0);
    gemm(xB, wvT + wo, bv + (size_t)l * DD, nullptr, vT, DD, DD, 0, 1);
    // fused attention -> yF
    attention_kernel<<<dim3(SS / 64, HH, BB), dim3(128), attnSmem, stream>>>(
        qB, kB, vT, yF, 0.125f /* DH^-0.5 */);
    // x = LN(x + y)
    add_layernorm<<<dim3(M), dim3(256), 0, stream>>>(xF, yF, xF, xB);
    // FFN
    gemm(xB, w1T + w1o, b1 + (size_t)l * DFF, nullptr, h1B, DFF, DD, 1, 0);   // ReLU
    gemm(h1B, w2T + w2o, b2 + (size_t)l * DD, yF, nullptr, DD, DFF, 0, 0);
    add_layernorm<<<dim3(M), dim3(256), 0, stream>>>(xF, yF, xF, xB);
  }

  // logits + softmax
  gemm(xB, woutT, bout, logit, nullptr, VV, DD, 0, 0);
  softmax_out<<<dim3(M), dim3(256), 0, stream>>>(logit, out);
}
